// FNO_23046794510986
// MI455X (gfx1250) — compile-verified, matrix operand order checked by reading, not run on hardware
//
#include <hip/hip_runtime.h>
#include <hip/hip_bf16.h>

// ---------------- geometry ----------------
#define NMODES 256
#define NLAYERS 4
#define TM 128            // rows per block (8 waves x 16 rows)
#define HPAD 264          // padded row stride (bf16 elems), 528B = 33*16B -> aligned + conflict-free
#define MPAD 264
#define MUPAD 72          // 144B rows, 16B aligned
#define W1PAD 72

// workspace offsets (bytes)
#define WS_WT   0                       // 4*256*256 bf16 (circulant, n-major / k-contiguous)
#define WS_W1T  524288                  // 256*64 bf16 (W1 transposed, k-contiguous)
#define WS_G3   557056                  // 4*3*256 f32 (rank-3 x-path)

#define SMEM_MAIN ((2*TM*HPAD + 256*MPAD)*2 + (TM*4 + 3*256 + 256)*4)  // = 276480 B

typedef __attribute__((ext_vector_type(16))) __bf16 v16bf;
typedef __attribute__((ext_vector_type(8)))  float  v8f;

// low 32 bits of a generic pointer into LDS == byte offset within the workgroup's LDS
__device__ __forceinline__ unsigned lds_off32(const void* p) {
  return (unsigned)(unsigned long long)p;
}

// CDNA5 async copy: 16B global -> LDS, tracked by ASYNCcnt (GVS addressing mode)
__device__ __forceinline__ void async_ld16(unsigned lds, unsigned long long gbase,
                                           unsigned goff) {
  asm volatile("global_load_async_to_lds_b128 %0, %1, %2"
               :: "v"(lds), "v"(goff), "s"(gbase) : "memory");
}
__device__ __forceinline__ void wait_async0() {
  asm volatile("s_wait_asynccnt 0" ::: "memory");
}

// 16x32 bf16 fragment from LDS per CDNA5 WMMA layout (lanes 0-15 row=lane, half=lane>>4
// shifts K by +8; VGPR0..3 = K{0..7}, VGPR4..7 = K{16..23}).  Two 16B-aligned b128 loads.
__device__ __forceinline__ v16bf load_frag(const __bf16* p, int rowStride, int row,
                                           int kk, int half) {
  union { v16bf v; uint4 q[2]; } f;
  const __bf16* rp = p + row * rowStride + kk + half * 8;
  f.q[0] = *(const uint4*)rp;
  f.q[1] = *(const uint4*)(rp + 16);
  return f.v;
}

// ------------------------------------------------------------------
// Setup: irfft(fw) -> circulant weights (bf16, K-major), lw folded on diagonal,
// rank-3 x-path table, W1 transposed to K-major bf16.
// ------------------------------------------------------------------
__global__ void fno_setup(const float* __restrict__ W1, const float* __restrict__ fw,
                          const float* __restrict__ lw, void* __restrict__ ws) {
  __bf16* Wt  = (__bf16*)((char*)ws + WS_WT);
  __bf16* W1t = (__bf16*)((char*)ws + WS_W1T);
  float*  G3  = (float*) ((char*)ws + WS_G3);

  int bid = blockIdx.x;
  if (bid < NLAYERS) {
    const int l = bid;
    __shared__ float g[NMODES];
    for (int n = threadIdx.x; n < NMODES; n += blockDim.x) {
      float s = fw[l * 129 + 0];
      for (int k = 1; k < 128; ++k) {
        int ph = (k * n) & 255;
        float ang = (float)ph * (6.2831853071795864f / 256.0f);
        s += 2.0f * fw[l * 129 + k] * __cosf(ang);
      }
      s += fw[l * 129 + 128] * ((n & 1) ? -1.0f : 1.0f);
      g[n] = s * (1.0f / 256.0f);
    }
    __syncthreads();
    for (int i = threadIdx.x; i < 3 * NMODES; i += blockDim.x) {
      int j = i >> 8, n = i & 255;
      G3[l * 3 * NMODES + i] = g[(n - j) & 255];
    }
    for (int i = threadIdx.x; i < NMODES * NMODES; i += blockDim.x) {
      int n = i >> 8, k = i & 255;
      float w = (k <= 252) ? g[(n - k - 3) & 255] : 0.0f;
      if (k == n) w += lw[l * NMODES + n];
      Wt[(size_t)l * NMODES * NMODES + i] = (__bf16)w;
    }
  } else {
    for (int i = threadIdx.x; i < NMODES * 64; i += blockDim.x) {
      int n = i >> 6, k = i & 63;
      W1t[i] = (__bf16)W1[k * NMODES + n];
    }
  }
}

// ------------------------------------------------------------------
// Main fused kernel: encoder GEMM + 4 circulant layers + decoder
// ------------------------------------------------------------------
__global__ void fno_main(const float* __restrict__ mu, const float* __restrict__ x,
                         const float* __restrict__ b1, const void* __restrict__ ws,
                         const float* __restrict__ W2, const float* __restrict__ b2,
                         float* __restrict__ out) {
  const __bf16* WtG  = (const __bf16*)((const char*)ws + WS_WT);
  const __bf16* W1tG = (const __bf16*)((const char*)ws + WS_W1T);
  const float*  G3G  = (const float*) ((const char*)ws + WS_G3);

  extern __shared__ uint4 smem4[];               // 16B-aligned dynamic LDS
  char* smem = (char*)smem4;
  __bf16* hA = (__bf16*)smem;                    // TM x HPAD
  __bf16* hB = hA + TM * HPAD;                   // TM x HPAD (mu staging uses stride MUPAD)
  __bf16* Mt = hB + TM * HPAD;                   // 256 x MPAD (W1t staging uses stride W1PAD)
  float*  xs  = (float*)(Mt + 256 * MPAD);       // TM x 4
  float*  gs  = xs + TM * 4;                     // 3 x 256
  float*  w2s = gs + 3 * 256;                    // 256

  const int tid  = threadIdx.x;
  const int lane = tid & 31;
  const int wave = tid >> 5;
  const int half = lane >> 4;
  const int rowbase = wave * 16;
  const int rb = blockIdx.x * TM;

  // ---- stage W1t via async copy unit; mu/x/W2 via regular loads ----
  {
    const unsigned long long gb = (unsigned long long)W1tG;
    const unsigned mtb = lds_off32(Mt);
    for (int idx = tid; idx < 256 * 8; idx += 256) {      // 256 rows x 8 x 16B
      int n = idx >> 3, c = idx & 7;
      async_ld16(mtb + (unsigned)(n * W1PAD + c * 8) * 2u, gb, (unsigned)idx * 16u);
    }
    for (int i = tid; i < TM * 64; i += 256) {
      int r = i >> 6, k = i & 63;
      hB[r * MUPAD + k] = (__bf16)mu[(size_t)(rb + r) * 64 + k];
    }
    for (int i = tid; i < TM * 3; i += 256) {
      int r = i / 3, c = i - r * 3;
      xs[r * 4 + c] = x[(size_t)(rb + r) * 3 + c];
    }
    w2s[tid] = W2[tid];
  }
  wait_async0();
  __syncthreads();

  // ---- encoder: h = relu(mu @ W1 + b1), K = 64 ----
  {
    const int rowlane = rowbase + (lane & 15);
    v16bf a0 = load_frag(hB, MUPAD, rowlane, 0,  half);
    v16bf a1 = load_frag(hB, MUPAD, rowlane, 32, half);
    for (int nt = 0; nt < 16; ++nt) {
      v8f acc = {0.f, 0.f, 0.f, 0.f, 0.f, 0.f, 0.f, 0.f};
      v16bf b0 = load_frag(Mt, W1PAD, nt * 16 + (lane & 15), 0, half);
      acc = __builtin_amdgcn_wmma_f32_16x16x32_bf16(false, a0, false, b0,
                                                    (short)0, acc, false, false);
      v16bf b1f = load_frag(Mt, W1PAD, nt * 16 + (lane & 15), 32, half);
      acc = __builtin_amdgcn_wmma_f32_16x16x32_bf16(false, a1, false, b1f,
                                                    (short)0, acc, false, false);
      const int N = nt * 16 + (lane & 15);
      const float bias = b1[N];
#pragma unroll
      for (int v = 0; v < 8; ++v) {
        int r = rowbase + v + 8 * half;
        hA[r * HPAD + N] = (__bf16)fmaxf(acc[v] + bias, 0.0f);
      }
    }
  }

  // ---- 4 circulant layers: h = relu(h @ Mt_l + x @ G3_l) ----
  __bf16* cur = hA;
  __bf16* nxt = hB;
  for (int l = 0; l < NLAYERS; ++l) {
    __syncthreads();
    {   // async-stage Mt_l (256 rows x 32 x 16B) + G3_l
      const unsigned long long gb = (unsigned long long)(WtG + (size_t)l * NMODES * NMODES);
      const unsigned mtb = lds_off32(Mt);
      for (int idx = tid; idx < 256 * 32; idx += 256) {
        int n = idx >> 5, c = idx & 31;
        async_ld16(mtb + (unsigned)(n * MPAD + c * 8) * 2u, gb, (unsigned)idx * 16u);
      }
      for (int i = tid; i < 3 * 256; i += 256) gs[i] = G3G[l * 3 * 256 + i];
    }
    if (l + 1 < NLAYERS)   // warm L2 for next layer's weights
      __builtin_prefetch(WtG + (size_t)(l + 1) * NMODES * NMODES + (size_t)tid * 256, 0, 1);
    wait_async0();
    __syncthreads();

    const int rowlane = rowbase + (lane & 15);
    v16bf afr[8];                                  // whole K-strip of A, reused for all N-tiles
#pragma unroll
    for (int t = 0; t < 8; ++t) afr[t] = load_frag(cur, HPAD, rowlane, t * 32, half);

    for (int nt = 0; nt < 16; ++nt) {
      v8f acc = {0.f, 0.f, 0.f, 0.f, 0.f, 0.f, 0.f, 0.f};
#pragma unroll
      for (int kk = 0; kk < 8; ++kk) {
        v16bf bm = load_frag(Mt, MPAD, nt * 16 + (lane & 15), kk * 32, half);
        acc = __builtin_amdgcn_wmma_f32_16x16x32_bf16(false, afr[kk], false, bm,
                                                      (short)0, acc, false, false);
      }
      const int N = nt * 16 + (lane & 15);
      const float g0 = gs[N], g1 = gs[256 + N], g2 = gs[512 + N];
#pragma unroll
      for (int v = 0; v < 8; ++v) {
        int r = rowbase + v + 8 * half;
        float xc = xs[r * 4 + 0] * g0 + xs[r * 4 + 1] * g1 + xs[r * 4 + 2] * g2;
        nxt[r * HPAD + N] = (__bf16)fmaxf(acc[v] + xc, 0.0f);
      }
    }
    __bf16* t = cur; cur = nxt; nxt = t;
  }

  // ---- decoder: out = h @ W2 + b2 (vectorized b128 LDS reads) ----
  __syncthreads();
  if (tid < TM) {
    const __bf16* hr = cur + tid * HPAD;
    float s = b2[0];
    for (int n0 = 0; n0 < NMODES; n0 += 8) {
      union { uint4 q; __bf16 b[8]; } u;
      u.q = *(const uint4*)(hr + n0);
#pragma unroll
      for (int j = 0; j < 8; ++j) s += (float)u.b[j] * w2s[n0 + j];
    }
    out[(size_t)rb + tid] = s;
  }
}

// ------------------------------------------------------------------
extern "C" void kernel_launch(void* const* d_in, const int* in_sizes, int n_in,
                              void* d_out, int out_size, void* d_ws, size_t ws_size,
                              hipStream_t stream) {
  const float* mu = (const float*)d_in[0];
  const float* x  = (const float*)d_in[1];
  const float* W1 = (const float*)d_in[2];
  const float* b1 = (const float*)d_in[3];
  const float* fw = (const float*)d_in[4];
  const float* lw = (const float*)d_in[5];
  const float* W2 = (const float*)d_in[6];
  const float* b2 = (const float*)d_in[7];
  float* out = (float*)d_out;

  const int B = in_sizes[0] / 64;

  fno_setup<<<NLAYERS + 1, 256, 0, stream>>>(W1, fw, lw, d_ws);
  fno_main<<<B / TM, 256, SMEM_MAIN, stream>>>(mu, x, b1, d_ws, W2, b2, out);
}
